// LocalDownSample_34368328302771
// MI455X (gfx1250) — matrix operand order repaired and checked by reading, not compiled, not run on hardware
//
#include <hip/hip_runtime.h>
#include <hip/hip_bf16.h>
#include <math.h>

// Problem dims (from setup_inputs): B=8, C=128, N=2048, K=32, M=1024
#define BB   8
#define CC   128
#define NN   2048
#define KK   32
#define MM   1024
#define PP   (BB * NN)   // 16384 points total

typedef __attribute__((ext_vector_type(16))) _Float16 v16h;
typedef __attribute__((ext_vector_type(8)))  float    v8f;

// ---------------- WMMA fragment loaders (ISA 7.12.2 layouts) ----------------
// A (16x32 f16, row-major source row of this lane's point):
//   lane half h = lane>>4 ; elements 0..7  <- k = 32*s + 8*h + i
//                           elements 8..15 <- k = 32*s + 16 + 8*h + i
__device__ __forceinline__ v16h load_frag_a(const _Float16* row, int s, int h) {
    v16h a;
    const _Float16* p0 = row + 32 * s + 8 * h;
    const _Float16* p1 = row + 32 * s + 16 + 8 * h;
#pragma unroll
    for (int i = 0; i < 8; ++i) { a[i] = p0[i]; a[8 + i] = p1[i]; }
    return a;
}
// B (32x16 f16). B[k][n] with n = lane&15; same k order as A: 32*s + 8*h + i, 32*s + 16 + 8*h + i.
__device__ __forceinline__ v16h load_frag_b(const _Float16* row, int s, int h) {
    v16h b;
    const _Float16* p0 = row + 32 * s + 8 * h;
    const _Float16* p1 = row + 32 * s + 16 + 8 * h;
#pragma unroll
    for (int i = 0; i < 8; ++i) { b[i] = p0[i]; b[8 + i] = p1[i]; }
    return b;
}

__device__ __forceinline__ v8f wmma_f16(v16h a, v16h b, v8f c) {
    return __builtin_amdgcn_wmma_f32_16x16x32_f16(false, a, false, b, (short)0, c, false, false);
}

// sorted-descending top-K insert, lexicographic (value desc, index asc) —
// matches jax.lax.top_k tie-breaking and is insertion-order independent.
__device__ __forceinline__ void insert_lex(float* tv, int* ti, float v, int idx) {
    if (v > tv[KK - 1] || (v == tv[KK - 1] && idx < ti[KK - 1])) {
        int pos = KK - 1;
        while (pos > 0 && (v > tv[pos - 1] || (v == tv[pos - 1] && idx < ti[pos - 1]))) {
            tv[pos] = tv[pos - 1];
            ti[pos] = ti[pos - 1];
            --pos;
        }
        tv[pos] = v;
        ti[pos] = idx;
    }
}

// ---------------- Kernel 0: Mt = (Wq^T Wk)^T in f16, Wv -> f16 ----------------
__global__ void prep_weights_kernel(const float* __restrict__ Wq,
                                    const float* __restrict__ Wk,
                                    const float* __restrict__ Wv,
                                    _Float16* __restrict__ mth,
                                    _Float16* __restrict__ wvh) {
    int co = blockIdx.x;      // column of M == row of Mt
    int c  = threadIdx.x;
    float s = 0.f;
    for (int o = 0; o < CC; ++o) s += Wq[o * CC + c] * Wk[o * CC + co];
    mth[co * CC + c] = (_Float16)s;
    wvh[co * CC + c] = (_Float16)Wv[co * CC + c];
}

// ---------------- Kernel 1: transpose x -> xt(f32)/xh(f16), sq = ||x_n||^2 ----
__global__ void transpose_kernel(const float* __restrict__ x,
                                 float* __restrict__ xt,
                                 _Float16* __restrict__ xh,
                                 float* __restrict__ sq) {
    int p = blockIdx.x;           // global point in [0, B*N)
    int b = p / NN, n = p % NN;
    int c = threadIdx.x;
    float v = x[((size_t)b * CC + c) * NN + n];
    xt[(size_t)p * CC + c] = v;
    xh[(size_t)p * CC + c] = (_Float16)v;
    __shared__ float red[CC];
    red[c] = v * v;
    __syncthreads();
    for (int s = CC / 2; s > 0; s >>= 1) {
        if (c < s) red[c] += red[c + s];
        __syncthreads();
    }
    if (c == 0) sq[p] = red[0];
}

// -------- Kernel 2: Gram matrix (WMMA) + top-32, wave-specialized pipeline ---
// 8 waves: waves 0-3 produce neg-dist strips via WMMA into a double-buffered
// LDS strip; waves 4-7 run the streaming top-32 scan on the previous strip.
#define STRIP   256
#define SPAD    (STRIP + 1)       // LDS row pad: kills bank conflicts
#define NSTRIPS (NN / STRIP)      // 8
__global__ void __launch_bounds__(256, 1)
knn_kernel(const _Float16* __restrict__ xh,
           const float* __restrict__ sq,
           int* __restrict__ knn) {
    __shared__ float strip[2][16][SPAD];   // ~33 KB double-buffered strip
    __shared__ int   cidx[16][SPAD];       // ~16 KB candidate indices (endgame)
    __shared__ float sqrow[16];

    const int b   = blockIdx.y;
    const int r0  = blockIdx.x * 16;
    const int tid = threadIdx.x;           // 0..255
    const int w8  = tid >> 5;              // wave id 0..7
    const bool producer = (w8 < 4);

    if (tid < 16) sqrow[tid] = sq[b * NN + r0 + tid];
    __syncthreads();

    // ---- producer state (waves 0-3) ----
    const int lane = tid & 31;
    const int h    = lane >> 4;
    const int lm   = lane & 15;
    v16h a[4];
    if (producer) {
        const _Float16* arow = xh + ((size_t)b * NN + r0 + lm) * CC;
#pragma unroll
        for (int s = 0; s < 4; ++s) a[s] = load_frag_a(arow, s, h);
    }

    // ---- scanner state (waves 4-7): 8 sub-scanners per row ----
    const int stid = tid - 128;            // 0..127 when scanner
    const int srow = stid >> 3;
    const int ssub = stid & 7;
    float tv[KK];
    int   ti[KK];
#pragma unroll
    for (int j = 0; j < KK; ++j) { tv[j] = -3.0e38f; ti[j] = 0x7fffffff; }

    // ---- software pipeline over strips ----
    for (int it = 0; it <= NSTRIPS; ++it) {
        if (producer && it < NSTRIPS) {
            const int c0 = it * STRIP;
            float (*sbuf)[SPAD] = strip[it & 1];
            for (int t = w8; t < STRIP / 16; t += 4) {
                const int cb = c0 + t * 16;
                const _Float16* brow = xh + ((size_t)b * NN + cb + lm) * CC;
                const float sqc = sq[b * NN + cb + lm];
                v16h bf[4];
#pragma unroll
                for (int s = 0; s < 4; ++s) bf[s] = load_frag_b(brow, s, h);
                v8f acc = {0.f, 0.f, 0.f, 0.f, 0.f, 0.f, 0.f, 0.f};
#pragma unroll
                for (int s = 0; s < 4; ++s) acc = wmma_f16(a[s], bf[s], acc);
#pragma unroll
                for (int r = 0; r < 8; ++r) {
                    const int m = r + 8 * h;
                    sbuf[m][t * 16 + lm] = 2.f * acc[r] - sqrow[m] - sqc;
                }
            }
        }
        if (!producer && it >= 1) {
            const int c0 = (it - 1) * STRIP;
            const float (*sbuf)[SPAD] = strip[(it - 1) & 1];
            for (int j = 0; j < 32; ++j) {
                const int col = ssub * 32 + j;
                insert_lex(tv, ti, sbuf[srow][col], c0 + col);
            }
        }
        __syncthreads();
    }

    // ---- dump 8x32 candidates per row, then merge ----
    if (!producer) {
        for (int j = 0; j < KK; ++j) {
            strip[0][srow][ssub * KK + j] = tv[j];
            cidx[srow][ssub * KK + j]     = ti[j];
        }
    }
    __syncthreads();
    if (tid < 16) {   // merge 256 pre-sorted candidates -> final top-32
        const int row = tid;
        float fv[KK];
        int   fi[KK];
#pragma unroll
        for (int j = 0; j < KK; ++j) { fv[j] = -3.0e38f; fi[j] = 0x7fffffff; }
        for (int cnd = 0; cnd < 8 * KK; ++cnd)
            insert_lex(fv, fi, strip[0][row][cnd], cidx[row][cnd]);
        int* o = knn + (size_t)(b * NN + r0 + row) * KK;
        for (int j = 0; j < KK; ++j) o[j] = fi[j];
    }
}

// ---------------- Kernel 3: g = xh @ Mt^T  (WMMA, 16 pts x 128 ch / block) ---
__global__ void __launch_bounds__(128, 1)
proj_g_kernel(const _Float16* __restrict__ xh,
              const _Float16* __restrict__ mth,
              float* __restrict__ g) {
    const int p0   = blockIdx.x * 16;         // global point base over B*N
    const int lane = threadIdx.x & 31;
    const int wave = threadIdx.x >> 5;
    const int h    = lane >> 4;
    const int lm   = lane & 15;

    const _Float16* arow = xh + (size_t)(p0 + lm) * CC;
    v16h a[4];
#pragma unroll
    for (int s = 0; s < 4; ++s) a[s] = load_frag_a(arow, s, h);

    for (int t = wave; t < CC / 16; t += 4) {
        const _Float16* brow = mth + (size_t)(t * 16 + lm) * CC;
        v16h bf[4];
#pragma unroll
        for (int s = 0; s < 4; ++s) bf[s] = load_frag_b(brow, s, h);
        v8f acc = {0.f, 0.f, 0.f, 0.f, 0.f, 0.f, 0.f, 0.f};
#pragma unroll
        for (int s = 0; s < 4; ++s) acc = wmma_f16(a[s], bf[s], acc);
#pragma unroll
        for (int r = 0; r < 8; ++r)
            g[(size_t)(p0 + r + 8 * h) * CC + t * 16 + lm] = acc[r];
    }
}

// ---------------- Kernel 4: energy -> softmax -> std (one wave / point) ------
__global__ void attn_kernel(const float* __restrict__ g,
                            const float* __restrict__ xt,
                            const int* __restrict__ knn,
                            float* __restrict__ att,
                            float* __restrict__ sel) {
    const int p = blockIdx.x;             // global point
    const int b = p / NN;
    const int j = threadIdx.x;            // 0..31

    const int idx = knn[(size_t)p * KK + j];
    const float4* gp = (const float4*)(g  + (size_t)p * CC);
    const float4* xn = (const float4*)(xt + (size_t)p * CC);
    const float4* xj = (const float4*)(xt + ((size_t)b * NN + idx) * CC);

    float e = 0.f;
#pragma unroll 8
    for (int i = 0; i < CC / 4; ++i) {
        const float4 gv = gp[i], nv = xn[i], jv = xj[i];
        e += gv.x * (jv.x - nv.x) + gv.y * (jv.y - nv.y) +
             gv.z * (jv.z - nv.z) + gv.w * (jv.w - nv.w);
    }
    e *= 0.08838834764831845f;            // 1/sqrt(128)

    float mx = e;
    for (int off = 16; off >= 1; off >>= 1) mx = fmaxf(mx, __shfl_xor(mx, off, 32));
    float ex = expf(e - mx);
    float sm = ex;
    for (int off = 16; off >= 1; off >>= 1) sm += __shfl_xor(sm, off, 32);
    const float aw = ex / sm;
    att[(size_t)p * KK + j] = aw;

    float s1 = aw;
    for (int off = 16; off >= 1; off >>= 1) s1 += __shfl_xor(s1, off, 32);
    const float mean = s1 * (1.f / KK);
    float d = aw - mean;
    float s2 = d * d;
    for (int off = 16; off >= 1; off >>= 1) s2 += __shfl_xor(s2, off, 32);
    if (j == 0) sel[p] = sqrtf(s2 * (1.f / KK));
}

// ---------------- Kernel 5: exact top-M via in-LDS bitonic sort (per batch) --
__global__ void topm_kernel(const float* __restrict__ sel,
                            int* __restrict__ topidx) {
    __shared__ float sv[NN];
    __shared__ int   si[NN];
    const int b = blockIdx.x;
    const int t = threadIdx.x;            // 1024 threads

    for (int i = t; i < NN; i += blockDim.x) { sv[i] = sel[b * NN + i]; si[i] = i; }
    __syncthreads();

    for (int k = 2; k <= NN; k <<= 1) {
        for (int js = k >> 1; js >= 1; js >>= 1) {
            const int i   = ((t / js) * (js << 1)) + (t % js);
            const int ixj = i + js;
            const bool dirDesc = ((i & k) == 0);
            // "i should precede ixj in descending (value desc, index asc)"
            const bool agt = (sv[i] > sv[ixj]) ||
                             (sv[i] == sv[ixj] && si[i] < si[ixj]);
            if (dirDesc ? !agt : agt) {
                float fv = sv[i]; sv[i] = sv[ixj]; sv[ixj] = fv;
                int   iv = si[i]; si[i] = si[ixj]; si[ixj] = iv;
            }
            __syncthreads();
        }
    }
    for (int i = t; i < MM; i += blockDim.x) topidx[b * MM + i] = si[i];
}

// ---------------- Kernel 6: u[m] = sum_j s_j (x_idx - x_n)  -> f16 -----------
__global__ void agg_kernel(const float* __restrict__ xt,
                           const float* __restrict__ att,
                           const int* __restrict__ knn,
                           const int* __restrict__ topidx,
                           _Float16* __restrict__ uh) {
    __shared__ float sa[KK];
    __shared__ int   sidx[KK];
    const int bm = blockIdx.x;            // 0 .. B*M-1
    const int b  = bm / MM;
    const int n  = topidx[bm];
    const size_t p = (size_t)b * NN + n;
    if (threadIdx.x < KK) {
        sa[threadIdx.x]   = att[p * KK + threadIdx.x];
        sidx[threadIdx.x] = knn[p * KK + threadIdx.x];
    }
    __syncthreads();
    const int c = threadIdx.x;
    const float xn = xt[p * CC + c];
    float u = 0.f;
#pragma unroll
    for (int j = 0; j < KK; ++j)
        u += sa[j] * (xt[((size_t)b * NN + sidx[j]) * CC + c] - xn);
    uh[(size_t)bm * CC + c] = (_Float16)u;
}

// ---------------- Kernel 7: out[b,c,m] = (u @ Wv^T) (WMMA) -------------------
__global__ void __launch_bounds__(128, 1)
out_kernel(const _Float16* __restrict__ uh,
           const _Float16* __restrict__ wvh,
           float* __restrict__ out) {
    const int p0   = blockIdx.x * 16;     // global selected-point base (B*M)
    const int b    = p0 / MM;             // M % 16 == 0 -> uniform batch
    const int m0   = p0 % MM;
    const int lane = threadIdx.x & 31;
    const int wave = threadIdx.x >> 5;
    const int h    = lane >> 4;
    const int lm   = lane & 15;

    const _Float16* arow = uh + (size_t)(p0 + lm) * CC;
    v16h a[4];
#pragma unroll
    for (int s = 0; s < 4; ++s) a[s] = load_frag_a(arow, s, h);

    for (int t = wave; t < CC / 16; t += 4) {
        const int co = t * 16 + lm;                    // output channel
        const _Float16* brow = wvh + (size_t)co * CC;  // row o of Wv
        v16h bf[4];
#pragma unroll
        for (int s = 0; s < 4; ++s) bf[s] = load_frag_b(brow, s, h);
        v8f acc = {0.f, 0.f, 0.f, 0.f, 0.f, 0.f, 0.f, 0.f};
#pragma unroll
        for (int s = 0; s < 4; ++s) acc = wmma_f16(a[s], bf[s], acc);
#pragma unroll
        for (int r = 0; r < 8; ++r) {
            const int m = m0 + r + 8 * h;
            out[((size_t)b * CC + co) * MM + m] = acc[r];
        }
    }
}

// ---------------- Host launch ------------------------------------------------
extern "C" void kernel_launch(void* const* d_in, const int* in_sizes, int n_in,
                              void* d_out, int out_size, void* d_ws, size_t ws_size,
                              hipStream_t stream) {
    const float* x  = (const float*)d_in[0];
    const float* Wq = (const float*)d_in[1];
    const float* Wk = (const float*)d_in[2];
    const float* Wv = (const float*)d_in[3];
    // d_in[4] = npts_ds (device scalar); value 1024 fixed at compile time (MM).
    float* out = (float*)d_out;
    char*  ws  = (char*)d_ws;

    // workspace carve-up (256B aligned)
    size_t off = 0;
    auto carve = [&](size_t bytes) { size_t o = off; off += (bytes + 255) & ~(size_t)255; return o; };
    float*    xt   = (float*)   (ws + carve((size_t)PP * CC * 4));
    _Float16* xh   = (_Float16*)(ws + carve((size_t)PP * CC * 2));
    float*    sq   = (float*)   (ws + carve((size_t)PP * 4));
    _Float16* mth  = (_Float16*)(ws + carve((size_t)CC * CC * 2));
    _Float16* wvh  = (_Float16*)(ws + carve((size_t)CC * CC * 2));
    float*    g    = (float*)   (ws + carve((size_t)PP * CC * 4));
    int*      knn  = (int*)     (ws + carve((size_t)PP * KK * 4));
    float*    att  = (float*)   (ws + carve((size_t)PP * KK * 4));
    float*    sel  = (float*)   (ws + carve((size_t)PP * 4));
    int*      tidx = (int*)     (ws + carve((size_t)BB * MM * 4));
    _Float16* uh   = (_Float16*)(ws + carve((size_t)BB * MM * CC * 2));

    prep_weights_kernel<<<dim3(CC), dim3(CC), 0, stream>>>(Wq, Wk, Wv, mth, wvh);
    transpose_kernel  <<<dim3(PP), dim3(CC), 0, stream>>>(x, xt, xh, sq);
    knn_kernel        <<<dim3(NN / 16, BB), dim3(256), 0, stream>>>(xh, sq, knn);
    proj_g_kernel     <<<dim3(PP / 16), dim3(128), 0, stream>>>(xh, mth, g);
    attn_kernel       <<<dim3(PP), dim3(KK), 0, stream>>>(g, xt, knn, att, sel);
    topm_kernel       <<<dim3(BB), dim3(1024), 0, stream>>>(sel, tidx);
    agg_kernel        <<<dim3(BB * MM), dim3(CC), 0, stream>>>(xt, att, knn, tidx, uh);
    out_kernel        <<<dim3(BB * MM / 16), dim3(128), 0, stream>>>(uh, wvh, out);
}
